// Denoising_Unet_1219770712800
// MI455X (gfx1250) — matrix operand order repaired and checked by reading, not compiled, not run on hardware
//
#include <hip/hip_runtime.h>

// ---------------------------------------------------------------------------
// Graph U-Net (MLP + GAT) for MI455X / gfx1250, wave32.
// - All dense layers: v_wmma_f32_16x16x32_f16, f16 activations/weights,
//   fp32 accumulate, fused bias/residual/PReLU epilogue.
// - Tile staging: global_load_async_to_lds_b128 (ASYNCcnt), double-buffered,
//   overlapping global->LDS DMA with WMMA compute.
// - Graph edge-softmax: ordered-uint atomicMax segment-max + f32 atomic adds.
// ---------------------------------------------------------------------------

typedef __attribute__((ext_vector_type(16))) _Float16 v16h;
typedef __attribute__((ext_vector_type(8)))  float    v8f;

#define NNODES 100000
#define NEDGES 1600000

#define BM 128
#define BN 64
#define BK 32
#define SAS 48   // LDS row stride (halves): 96B rows -> 16B-aligned b128 chunks
#define SBS 48

// order-preserving float<->uint key for atomic segment-max
__device__ __forceinline__ unsigned f2key(float f) {
  int i = __float_as_int(f);
  return (i >= 0) ? ((unsigned)i | 0x80000000u) : ~(unsigned)i;
}
__device__ __forceinline__ float key2f(unsigned k) {
  int i = (k & 0x80000000u) ? (int)(k ^ 0x80000000u) : ~(int)k;
  return __int_as_float(i);
}

// async 16B global -> LDS copy (ASYNCcnt-tracked, no VGPR data path)
__device__ __forceinline__ void async_b128(unsigned lds_off, const void* gaddr) {
  asm volatile("global_load_async_to_lds_b128 %0, %1, off"
               :: "v"(lds_off), "v"(gaddr)
               : "memory");
}

// ---------------------------------------------------------------------------
// Fused GEMM: out = act( X @ Wt^T + bias [+ R] )
//   X  : [N,K]  f16, row stride ldX
//   Wt : [M,K]  f16 (pre-transposed weights, contiguous K)
//   Y  : [N,M]  f16 (optional), Y32 : [N,M] f32 (optional)
// 8 waves; each wave: 16 rows x 64 cols = 4 WMMA accumulators.
// ---------------------------------------------------------------------------
__global__ __launch_bounds__(256) void wmma_gemm_f16(
    const _Float16* __restrict__ X, int ldX,
    const _Float16* __restrict__ Wt,
    const float* __restrict__ bias,
    const _Float16* __restrict__ R, int ldR,
    _Float16* __restrict__ Y, int ldY,
    float* __restrict__ Y32, int ldY32,
    int N, int K, int M,
    const float* __restrict__ alpha_p)
{
  __shared__ _Float16 sA[2][BM * SAS];
  __shared__ _Float16 sB[2][BN * SBS];

  const int tid  = threadIdx.x;
  const int wid  = tid >> 5;
  const int lane = tid & 31;
  const int row0 = blockIdx.x * BM;
  const int col0 = blockIdx.y * BN;
  const bool full = (row0 + BM) <= N;   // block-uniform: no row guard needed

  v8f acc[4];
  for (int t = 0; t < 4; ++t)
    for (int r = 0; r < 8; ++r) acc[t][r] = 0.0f;

  // stage one BK-slice into LDS buffer b via async DMA.
  // per wave: exactly 3 async instructions (2 for A, 1 for B).
  auto stage = [&](int b, int k0) {
    if (full) {
      for (int p = 0; p < 2; ++p) {
        int c  = tid + (p << 8);        // 0..511 16B chunks of A (128x32 f16)
        int r  = c >> 2;
        int kc = (c & 3) << 3;          // 8 halves per chunk
        async_b128((unsigned)(size_t)&sA[b][r * SAS + kc],
                   X + (size_t)(row0 + r) * ldX + k0 + kc);
      }
    } else {
      for (int p = 0; p < 2; ++p) {
        int c    = tid + (p << 8);
        int r    = c >> 2;
        int kc   = (c & 3) << 3;
        int grow = row0 + r;
        if (grow < N)
          async_b128((unsigned)(size_t)&sA[b][r * SAS + kc],
                     X + (size_t)grow * ldX + k0 + kc);
      }
    }
    {
      int n  = tid >> 2;                // 0..63, 4 chunks per output col
      int kc = (tid & 3) << 3;
      async_b128((unsigned)(size_t)&sB[b][n * SBS + kc],
                 Wt + (size_t)(col0 + n) * K + k0 + kc);
    }
  };

  const int nk = K / BK;
  stage(0, 0);
  for (int kt = 0; kt < nk; ++kt) {
    const int cur = kt & 1;
    if (kt + 1 < nk) {
      stage(cur ^ 1, (kt + 1) * BK);                    // prefetch next slice
      asm volatile("s_wait_asynccnt 0x3" ::: "memory"); // current slice done
    } else {
      asm volatile("s_wait_asynccnt 0x0" ::: "memory");
    }
    __syncthreads();

    // preload A + all 4 B fragments, then issue 4 WMMAs back-to-back
    const int arow = wid * 16 + (lane & 15);
    const int akb  = (lane < 16) ? 0 : 8;
    const int bkb  = (lane < 16) ? 0 : 8;
    union { v16h v; uint4 q[2]; } af, bf[4];
    af.q[0] = *(const uint4*)&sA[cur][arow * SAS + akb];
    af.q[1] = *(const uint4*)&sA[cur][arow * SAS + akb + 16];
    for (int t = 0; t < 4; ++t) {
      int n = t * 16 + (lane & 15);
      bf[t].q[0] = *(const uint4*)&sB[cur][n * SBS + bkb];
      bf[t].q[1] = *(const uint4*)&sB[cur][n * SBS + bkb + 16];
    }
    for (int t = 0; t < 4; ++t)
      acc[t] = __builtin_amdgcn_wmma_f32_16x16x32_f16(
          false, af.v, false, bf[t].v, (short)0, acc[t], false, false);
    __syncthreads();
  }

  // fused epilogue: bias + residual + PReLU, dual-precision store
  const float alpha = alpha_p ? alpha_p[0] : 0.0f;
  const int mofs = (lane < 16) ? 0 : 8;
  for (int t = 0; t < 4; ++t) {
    int col  = col0 + t * 16 + (lane & 15);
    float bv = bias ? bias[col] : 0.0f;
    for (int r = 0; r < 8; ++r) {
      int grow = row0 + wid * 16 + mofs + r;
      if (grow < N) {
        float v = acc[t][r] + bv;
        if (R) v += (float)R[(size_t)grow * ldR + col];
        if (alpha_p) v = (v >= 0.0f) ? v : alpha * v;
        if (Y)   Y[(size_t)grow * ldY + col] = (_Float16)v;
        if (Y32) Y32[(size_t)grow * ldY32 + col] = v;
      }
    }
  }
}

// ---------------------------------------------------------------------------
// helpers: conversions, adds, weight prep
// ---------------------------------------------------------------------------
__global__ void cvt_f32_to_f16(const float* __restrict__ x, int ldx,
                               _Float16* __restrict__ y, int N, int C) {
  int i = blockIdx.x * blockDim.x + threadIdx.x;
  if (i >= N * C) return;
  int n = i / C, c = i - n * C;
  y[i] = (_Float16)x[(size_t)n * ldx + c];
}

__global__ void add_f16(const _Float16* __restrict__ a, int lda,
                        const _Float16* __restrict__ b, int ldb,
                        _Float16* __restrict__ z, int N, int C) {
  int i = blockIdx.x * blockDim.x + threadIdx.x;
  if (i >= N * C) return;
  int n = i / C, c = i - n * C;
  z[i] = (_Float16)((float)a[(size_t)n * lda + c] + (float)b[(size_t)n * ldb + c]);
}

// W[K,M] f32 -> Wt[M,K] f16 (transposed, contiguous K for async B staging)
__global__ void wt_prep(const float* __restrict__ W, _Float16* __restrict__ Wt,
                        int K, int M) {
  int i = blockIdx.x * blockDim.x + threadIdx.x;
  if (i >= K * M) return;
  int m = i / K, k = i - m * K;
  Wt[i] = (_Float16)W[(size_t)k * M + m];
}

// ---------------------------------------------------------------------------
// LayerNorm + PReLU on f16 rows, in place. One wave32 per row.
// ---------------------------------------------------------------------------
__global__ __launch_bounds__(256) void layernorm_prelu(
    _Float16* __restrict__ Xb, int ld, int N, int M,
    const float* __restrict__ g, const float* __restrict__ be,
    const float* __restrict__ alpha_p)
{
  int row  = blockIdx.x * 8 + (threadIdx.x >> 5);
  int lane = threadIdx.x & 31;
  if (row >= N) return;
  _Float16* x = Xb + (size_t)row * ld;
  float v[8];
  int nch = M >> 5;
  float s = 0.f, s2 = 0.f;
  for (int i = 0; i < nch; ++i) {
    float t = (float)x[lane + (i << 5)];
    v[i] = t; s += t; s2 += t * t;
  }
  for (int m = 16; m >= 1; m >>= 1) {
    s  += __shfl_xor(s,  m, 32);
    s2 += __shfl_xor(s2, m, 32);
  }
  float mu   = s / (float)M;
  float var  = s2 / (float)M - mu * mu;
  float rstd = rsqrtf(var + 1e-5f);
  float a    = alpha_p[0];
  for (int i = 0; i < nch; ++i) {
    int c = lane + (i << 5);
    float t = (v[i] - mu) * rstd * g[c] + be[c];
    x[c] = (_Float16)((t >= 0.f) ? t : a * t);
  }
}

// ---------------------------------------------------------------------------
// GAT pieces
// ---------------------------------------------------------------------------
__global__ __launch_bounds__(256) void gat_attn(
    const _Float16* __restrict__ feat, int N, int H, int F,
    const float* __restrict__ al, const float* __restrict__ ar,
    float* __restrict__ el, float* __restrict__ er)
{
  int n    = blockIdx.x * 8 + (threadIdx.x >> 5);
  int lane = threadIdx.x & 31;
  if (n >= N) return;
  const _Float16* f = feat + (size_t)n * H * F;
  for (int h = 0; h < H; ++h) {
    float sl = 0.f, sr = 0.f;
    for (int c = lane; c < F; c += 32) {
      float t = (float)f[h * F + c];
      sl += t * al[h * F + c];
      sr += t * ar[h * F + c];
    }
    for (int m = 16; m >= 1; m >>= 1) {
      sl += __shfl_xor(sl, m, 32);
      sr += __shfl_xor(sr, m, 32);
    }
    if (lane == 0) { el[n * H + h] = sl; er[n * H + h] = sr; }
  }
}

__global__ void gat_node_init(unsigned* __restrict__ ekey,
                              float* __restrict__ den, int n) {
  int i = blockIdx.x * blockDim.x + threadIdx.x;
  if (i < n) { ekey[i] = 0u; den[i] = 0.0f; }
}

__global__ void edge_logits_max(const float* __restrict__ el,
                                const float* __restrict__ er,
                                const int* __restrict__ src,
                                const int* __restrict__ dst,
                                int E, int H,
                                float* __restrict__ ebuf,
                                unsigned* __restrict__ ekey)
{
  int i = blockIdx.x * blockDim.x + threadIdx.x;
  if (i >= E * H) return;
  int e = i / H, h = i - e * H;
  float v = el[src[e] * H + h] + er[dst[e] * H + h];
  v = (v >= 0.f) ? v : 0.2f * v;            // leaky_relu
  ebuf[i] = v;
  atomicMax(&ekey[dst[e] * H + h], f2key(v));
}

__global__ void emax_decode(const unsigned* __restrict__ ekey,
                            float* __restrict__ emax, int n) {
  int i = blockIdx.x * blockDim.x + threadIdx.x;
  if (i >= n) return;
  float f = key2f(ekey[i]);
  emax[i] = (f == f && f < 1e38f && f > -1e38f) ? f : 0.0f;
}

__global__ void edge_exp_sum(const float* __restrict__ emax,
                             const int* __restrict__ dst,
                             float* __restrict__ ebuf,
                             float* __restrict__ den, int E, int H)
{
  int i = blockIdx.x * blockDim.x + threadIdx.x;
  if (i >= E * H) return;
  int e = i / H, h = i - e * H;
  float ee = __expf(ebuf[i] - emax[dst[e] * H + h]);
  ebuf[i] = ee;
  atomicAdd(&den[dst[e] * H + h], ee);
}

__global__ void fill_bias(float* __restrict__ out, int ldO,
                          const float* __restrict__ b, int N, int HF) {
  int i = blockIdx.x * blockDim.x + threadIdx.x;
  if (i >= N * HF) return;
  int n = i / HF, c = i - n * HF;
  out[(size_t)n * ldO + c] = b[c];
}

__global__ void edge_scatter(const _Float16* __restrict__ feat,
                             const float* __restrict__ ebuf,
                             const float* __restrict__ den,
                             const int* __restrict__ src,
                             const int* __restrict__ dst,
                             float* __restrict__ out, int ldO,
                             int E, int H, int F)
{
  int i = blockIdx.x * blockDim.x + threadIdx.x;
  int HF = H * F;
  if (i >= E * HF) return;
  int e = i / HF, c = i - e * HF;
  int h = c / F;
  int d = dst[e];
  float a = ebuf[e * H + h] / fmaxf(den[d * H + h], 1e-9f);
  atomicAdd(&out[(size_t)d * ldO + c], (float)feat[(size_t)src[e] * HF + c] * a);
}

// ---------------------------------------------------------------------------
extern "C" void kernel_launch(void* const* d_in, const int* in_sizes, int n_in,
                              void* d_out, int out_size, void* d_ws, size_t ws_size,
                              hipStream_t stream)
{
  (void)in_sizes; (void)n_in; (void)out_size; (void)ws_size;
  const int N = NNODES, E = NEDGES;

  const float* x_t = (const float*)d_in[0];
  const int*   src = (const int*)d_in[2];
  const int*   dst = (const int*)d_in[3];

  // JAX pytree (sorted-key) flattening of params:
  const float* down0[4];  for (int i = 0; i < 4;  ++i) down0[i]  = (const float*)d_in[4  + i];
  const float* down1[4];  for (int i = 0; i < 4;  ++i) down1[i]  = (const float*)d_in[8  + i];
  const float* mlpin[12]; for (int i = 0; i < 12; ++i) mlpin[i]  = (const float*)d_in[12 + i];
  const float* mlpmid[12];for (int i = 0; i < 12; ++i) mlpmid[i] = (const float*)d_in[24 + i];
  const float* mlpout[12];for (int i = 0; i < 12; ++i) mlpout[i] = (const float*)d_in[36 + i];
  const float* up0[4];    for (int i = 0; i < 4;  ++i) up0[i]    = (const float*)d_in[48 + i];
  const float* up1[4];    for (int i = 0; i < 4;  ++i) up1[i]    = (const float*)d_in[52 + i];

  // ---- workspace carving ----
  char* wp = (char*)d_ws;
  auto carve = [&](size_t bytes) {
    char* p = wp; wp += (bytes + 255) & ~(size_t)255; return p;
  };
  _Float16* Xh  = (_Float16*)carve((size_t)N * 128 * 2);
  _Float16* T0  = (_Float16*)carve((size_t)N * 256 * 2);   // mlp_in t0 / slots
  _Float16* U   = (_Float16*)carve((size_t)N * 256 * 2);   // mlp_in u  / slots
  _Float16* Ch  = (_Float16*)carve((size_t)N * 128 * 2);
  _Float16* FE  = (_Float16*)carve((size_t)N * 128 * 2);   // GAT feat
  _Float16* AD  = (_Float16*)carve((size_t)N * 128 * 2);   // h + skip sums
  float*    G32 = (float*)   carve((size_t)N * 128 * 4);   // GAT f32 accum
  float*    EB  = (float*)   carve((size_t)E * 4 * 4);
  float*    EL  = (float*)   carve((size_t)N * 4 * 4);
  float*    ER  = (float*)   carve((size_t)N * 4 * 4);
  float*    EMX = (float*)   carve((size_t)N * 4 * 4);
  unsigned* EK  = (unsigned*)carve((size_t)N * 4 * 4);
  float*    DEN = (float*)   carve((size_t)N * 4 * 4);
  _Float16* WT  = (_Float16*)carve((size_t)512 * 1024 * 2); // transposed f16 weights

  _Float16* S0h = T0;
  _Float16* S1h = T0 + (size_t)N * 128;
  _Float16* S2h = U;
  _Float16* S3h = U  + (size_t)N * 128;

  float* out = (float*)d_out;                 // [N,128]
  float* OH  = out + (size_t)N * 128;         // [N,256] concat(out_hidden)

  // ---- one-time prep: x -> f16, weights -> f16 transposed ----
  cvt_f32_to_f16<<<dim3((N * 128 + 255) / 256), dim3(256), 0, stream>>>(x_t, 128, Xh, N, 128);

  _Float16* wtp = WT;
  auto prep = [&](const float* Wsrc, int K, int M) {
    _Float16* d = wtp; wtp += (size_t)K * M;
    wt_prep<<<dim3((K * M + 255) / 256), dim3(256), 0, stream>>>(Wsrc, d, K, M);
    return d;
  };
  struct MLPW { _Float16 *W1, *W2, *Wi, *Wo; };
  auto prep_mlp = [&](const float** P, int in_d, int hid, int out_d) {
    MLPW m;
    m.W1 = prep(P[0], hid,  hid);
    m.W2 = prep(P[1], hid,  hid);
    m.Wi = prep(P[2], in_d, hid);
    m.Wo = prep(P[3], hid,  out_d);
    return m;
  };
  MLPW Wmin = prep_mlp(mlpin,  128, 256, 128);
  MLPW Wmid = prep_mlp(mlpmid, 128, 128, 128);
  MLPW Wout = prep_mlp(mlpout, 128, 128, 128);
  _Float16* Wd0 = prep(down0[0], 128, 128);
  _Float16* Wd1 = prep(down1[0], 128, 128);
  _Float16* Wu0 = prep(up0[0],   128, 128);
  _Float16* Wu1 = prep(up1[0],   128, 128);

  auto gemm = [&](const _Float16* X, int ldX, const _Float16* Wt,
                  const float* bias, const _Float16* R, int ldR,
                  _Float16* Y, int ldY, float* Y32, int ldY32,
                  int K, int M, const float* alpha) {
    dim3 g((N + BM - 1) / BM, M / BN), b(256);
    wmma_gemm_f16<<<g, b, 0, stream>>>(X, ldX, Wt, bias, R, ldR,
                                       Y, ldY, Y32, ldY32, N, K, M, alpha);
  };

  // mlp sorted keys: 4=a1 5=ao 6=b1 7=b2 8=be 9=bi 10=bo 11=g
  auto run_mlp = [&](const _Float16* X, int ldX, int in_d, int hid, int out_d,
                     const float** P, const MLPW& Wm,
                     _Float16* t0, _Float16* u,
                     _Float16* Yh, int ldYh, float* Y32, int ldY32) {
    const float *a1 = P[4], *ao = P[5], *b1 = P[6], *b2 = P[7];
    const float *be = P[8], *bi = P[9], *bo = P[10], *g = P[11];
    gemm(X,  ldX, Wm.Wi, bi, nullptr, 0, t0, hid, nullptr, 0, in_d, hid, nullptr);
    gemm(t0, hid, Wm.W1, b1, nullptr, 0, u,  hid, nullptr, 0, hid,  hid, nullptr);
    layernorm_prelu<<<dim3((N + 7) / 8), dim3(256), 0, stream>>>(u, hid, N, hid, g, be, a1);
    gemm(u,  hid, Wm.W2, b2, t0, hid, t0, hid, nullptr, 0, hid, hid, nullptr);
    gemm(t0, hid, Wm.Wo, bo, nullptr, 0, Yh, ldYh, Y32, ldY32, hid, out_d, ao);
  };

  auto run_gat = [&](const _Float16* X, int ldX, const _Float16* Wt,
                     const float* al, const float* ar, const float* b,
                     int H, int F, float* O32, int ldO) {
    int HF = H * F;  // always 128
    gemm(X, ldX, Wt, nullptr, nullptr, 0, FE, HF, nullptr, 0, 128, HF, nullptr);
    gat_attn<<<dim3((N + 7) / 8), dim3(256), 0, stream>>>(FE, N, H, F, al, ar, EL, ER);
    gat_node_init<<<dim3((N * H + 255) / 256), dim3(256), 0, stream>>>(EK, DEN, N * H);
    edge_logits_max<<<dim3((E * H + 255) / 256), dim3(256), 0, stream>>>(EL, ER, src, dst, E, H, EB, EK);
    emax_decode<<<dim3((N * H + 255) / 256), dim3(256), 0, stream>>>(EK, EMX, N * H);
    edge_exp_sum<<<dim3((E * H + 255) / 256), dim3(256), 0, stream>>>(EMX, dst, EB, DEN, E, H);
    fill_bias<<<dim3((N * HF + 255) / 256), dim3(256), 0, stream>>>(O32, ldO, b, N, HF);
    edge_scatter<<<dim3((E * HF + 255) / 256), dim3(256), 0, stream>>>(FE, EB, DEN, src, dst, O32, ldO, E, H, F);
  };

  const int CV = (N * 128 + 255) / 256;

  // ---------------- pipeline ----------------
  run_mlp(Xh, 128, 128, 256, 128, mlpin, Wmin, T0, U, Ch, 128, nullptr, 0);

  run_gat(Ch, 128, Wd0, down0[1], down0[2], down0[3], 4, 32, G32, 128);
  cvt_f32_to_f16<<<dim3(CV), dim3(256), 0, stream>>>(G32, 128, S1h, N, 128);   // down_hidden0

  run_gat(S1h, 128, Wd1, down1[1], down1[2], down1[3], 4, 32, G32, 128);
  cvt_f32_to_f16<<<dim3(CV), dim3(256), 0, stream>>>(G32, 128, S3h, N, 128);   // down_hidden1

  run_mlp(S3h, 128, 128, 128, 128, mlpmid, Wmid, Ch, S0h, S2h, 128, nullptr, 0);

  add_f16<<<dim3(CV), dim3(256), 0, stream>>>(S2h, 128, S3h, 128, AD, N, 128); // h + dh1
  run_gat(AD, 128, Wu0, up0[1], up0[2], up0[3], 4, 32, OH, 256);               // out_hidden0
  cvt_f32_to_f16<<<dim3(CV), dim3(256), 0, stream>>>(OH, 256, Ch, N, 128);

  add_f16<<<dim3(CV), dim3(256), 0, stream>>>(Ch, 128, S1h, 128, AD, N, 128);  // h + dh0
  run_gat(AD, 128, Wu1, up1[1], up1[2], up1[3], 1, 128, OH + 128, 256);        // out_hidden1
  cvt_f32_to_f16<<<dim3(CV), dim3(256), 0, stream>>>(OH + 128, 256, S0h, N, 128);

  run_mlp(S0h, 128, 128, 128, 128, mlpout, Wout, Ch, S2h, nullptr, 0, out, 128);
}
